// DarkManifoldRealBiochemistry_74663711474155
// MI455X (gfx1250) — compile-verified, matrix operand order checked by reading, not run on hardware
//
#include <hip/hip_runtime.h>

typedef __attribute__((ext_vector_type(16))) _Float16 v16h;
typedef __attribute__((ext_vector_type(8)))  float    v8f;

#define N_METS  114
#define N_RXNS  64
#define N_GENES 212
#define DT_STEP 0.01f
#define EPS_C   1e-6f
#define WAVES_PER_BLOCK 8
#define BLOCK_THREADS   (WAVES_PER_BLOCK * 32)

#define KB_G 7    // ceil(224/32) K-blocks for genes
#define KB_M 4    // 128/32      K-blocks for metabolites
#define KB_R 2    // 64/32       K-blocks for reactions
#define FRAG 512  // 32 lanes * 16 halfs per swizzled fragment

// ---- f32 pair -> packed f16 pair (v_cvt_pk_rtz_f16_f32) ----
__device__ __forceinline__ unsigned pk2(float x, float y) {
  return __builtin_bit_cast(unsigned, __builtin_amdgcn_cvt_pkrtz(x, y));
}

struct P8 { unsigned w[8]; };

__device__ __forceinline__ v16h pack16(const float* f) {
  P8 s;
#pragma unroll
  for (int i = 0; i < 8; ++i) s.w[i] = pk2(f[2 * i], f[2 * i + 1]);
  return __builtin_bit_cast(v16h, s);
}

// ---- A fragment, full block: contiguous float2 loads (8B aligned) ----
__device__ __forceinline__ v16h a_frag_f2(const float* row, int kb, int h) {
  float f[16];
  const float2* q0 = (const float2*)(row + kb + h * 8);
  const float2* q1 = (const float2*)(row + kb + 16 + h * 8);
#pragma unroll
  for (int i = 0; i < 4; ++i) { float2 x = q0[i]; f[2*i] = x.x; f[2*i+1] = x.y; }
#pragma unroll
  for (int i = 0; i < 4; ++i) { float2 x = q1[i]; f[8+2*i] = x.x; f[9+2*i] = x.y; }
  return pack16(f);
}

// ---- A fragment, tail block: index-clamped (no exec divergence) ----
__device__ __forceinline__ v16h a_frag_clamp(const float* row, int kb, int h, int kmax) {
  float f[16];
#pragma unroll
  for (int i = 0; i < 8; ++i) {
    int k0 = kb + h * 8 + i;       k0 = (k0 < kmax) ? k0 : (kmax - 1);
    int k1 = kb + 16 + h * 8 + i;  k1 = (k1 < kmax) ? k1 : (kmax - 1);
    f[i] = row[k0];
    f[8 + i] = row[k1];
  }
  return pack16(f);
}

// ---- log(conc+eps) fragments ----
__device__ __forceinline__ v16h a_frag_log_f2(const float* row, int kb, int h) {
  float f[16];
  const float2* q0 = (const float2*)(row + kb + h * 8);
  const float2* q1 = (const float2*)(row + kb + 16 + h * 8);
#pragma unroll
  for (int i = 0; i < 4; ++i) { float2 x = q0[i]; f[2*i] = x.x; f[2*i+1] = x.y; }
#pragma unroll
  for (int i = 0; i < 4; ++i) { float2 x = q1[i]; f[8+2*i] = x.x; f[9+2*i] = x.y; }
#pragma unroll
  for (int i = 0; i < 16; ++i) f[i] = __logf(f[i] + EPS_C);
  return pack16(f);
}

__device__ __forceinline__ v16h a_frag_log_clamp(const float* row, int kb, int h, int kmax) {
  float f[16];
#pragma unroll
  for (int i = 0; i < 8; ++i) {
    int k0 = kb + h * 8 + i;       k0 = (k0 < kmax) ? k0 : (kmax - 1);
    int k1 = kb + 16 + h * 8 + i;  k1 = (k1 < kmax) ? k1 : (kmax - 1);
    f[i] = row[k0];
    f[8 + i] = row[k1];
  }
#pragma unroll
  for (int i = 0; i < 16; ++i) f[i] = __logf(f[i] + EPS_C);
  return pack16(f);
}

// ---- fragment from swizzled LDS: one 32B contiguous chunk per lane ----
__device__ __forceinline__ v16h frag_sw(const _Float16* sw) {
  const int L = threadIdx.x & 31;
  return *(const v16h*)(sw + L * 16);
}

#define WMMA_F16(A, Bb, C) \
  __builtin_amdgcn_wmma_f32_16x16x32_f16(false, (A), false, (Bb), (short)0, (C), false, false)

__device__ __forceinline__ v8f zero8() {
  v8f z;
#pragma unroll
  for (int i = 0; i < 8; ++i) z[i] = 0.0f;
  return z;
}

__global__ __launch_bounds__(BLOCK_THREADS)
void syn3a_kinetics_step(const float* __restrict__ conc,
                         const float* __restrict__ gene_expr,
                         const float* __restrict__ S,
                         const float* __restrict__ G,
                         const float* __restrict__ kcat,
                         float* __restrict__ out,
                         int n_tiles) {
  // B matrices stored pre-swizzled in WMMA-fragment order:
  //   [kblk][nb][lane(32)][elem(16)]  -> per-lane 32B contiguous
  __shared__ alignas(32) _Float16 G_sw[KB_G * 4 * FRAG];      // 28672 B
  __shared__ alignas(32) _Float16 sneg_sw[KB_M * 4 * FRAG];   // 16384 B
  __shared__ alignas(32) _Float16 st_sw[KB_R * 8 * FRAG];     // 16384 B
  __shared__ alignas(32) _Float16 rates_sw[WAVES_PER_BLOCK * KB_R * FRAG]; // 16384 B
  __shared__ float kcat_lds[N_RXNS];

  const int tid = threadIdx.x;

  // ---- stage graph matrices into swizzled LDS (f32 -> f16, zero-padded) ----
  for (int idx = tid; idx < KB_G * 4 * FRAG; idx += BLOCK_THREADS) {
    const int i = idx & 15, L = (idx >> 4) & 31, fi = idx >> 9;
    const int nb = fi & 3, kblk = fi >> 2, h = L >> 4;
    const int n = nb * 16 + (L & 15);
    const int k = kblk * 32 + ((i < 8) ? (h * 8 + i) : (16 + h * 8 + (i - 8)));
    G_sw[idx] = (_Float16)((k < N_GENES) ? G[k * N_RXNS + n] : 0.0f);
  }
  for (int idx = tid; idx < KB_M * 4 * FRAG; idx += BLOCK_THREADS) {
    const int i = idx & 15, L = (idx >> 4) & 31, fi = idx >> 9;
    const int nb = fi & 3, kblk = fi >> 2, h = L >> 4;
    const int n = nb * 16 + (L & 15);
    const int k = kblk * 32 + ((i < 8) ? (h * 8 + i) : (16 + h * 8 + (i - 8)));
    const float s = (k < N_METS) ? S[k * N_RXNS + n] : 0.0f;
    sneg_sw[idx] = (_Float16)fmaxf(-s, 0.0f);
  }
  for (int idx = tid; idx < KB_R * 8 * FRAG; idx += BLOCK_THREADS) {
    const int i = idx & 15, L = (idx >> 4) & 31, fi = idx >> 9;
    const int nb = fi & 7, kblk = fi >> 3, h = L >> 4;
    const int n = nb * 16 + (L & 15);  // metabolite index (N of S^T)
    const int k = kblk * 32 + ((i < 8) ? (h * 8 + i) : (16 + h * 8 + (i - 8))); // reaction
    st_sw[idx] = (_Float16)((n < N_METS) ? S[n * N_RXNS + k] : 0.0f);
  }
  if (tid < N_RXNS) kcat_lds[tid] = kcat[tid];
  __syncthreads();

  const int wave    = tid >> 5;
  const int lane    = tid & 31;
  const int m_a     = lane & 15;       // A-fragment row for this lane
  const int h_a     = lane >> 4;       // A-fragment half selector
  const int col     = lane & 15;       // C/D layout: N index
  const int row_off = (lane >> 4) * 8; // C/D layout: M offset
  _Float16* rs = rates_sw + wave * (KB_R * FRAG);

  for (int tile = blockIdx.x * WAVES_PER_BLOCK + wave; tile < n_tiles;
       tile += gridDim.x * WAVES_PER_BLOCK) {
    const size_t cell0 = (size_t)tile * 16;
    const float* ge_row = gene_expr + (cell0 + m_a) * (size_t)N_GENES;
    const float* c_row  = conc + (cell0 + m_a) * (size_t)N_METS;

    // ---- GEMM 1: enzyme = gene_expr @ G  ([16,224] x [224,64]) ----
    v8f accE[4];
#pragma unroll
    for (int nb = 0; nb < 4; ++nb) accE[nb] = zero8();
#pragma unroll
    for (int kblk = 0; kblk < KB_G - 1; ++kblk) {   // K 0..191: unconditional
      const v16h a = a_frag_f2(ge_row, kblk * 32, h_a);
#pragma unroll
      for (int nb = 0; nb < 4; ++nb)
        accE[nb] = WMMA_F16(a, frag_sw(G_sw + (kblk * 4 + nb) * FRAG), accE[nb]);
    }
    {                                                // K 192..223: clamped tail
      const v16h a = a_frag_clamp(ge_row, (KB_G - 1) * 32, h_a, N_GENES);
#pragma unroll
      for (int nb = 0; nb < 4; ++nb)
        accE[nb] = WMMA_F16(a, frag_sw(G_sw + ((KB_G - 1) * 4 + nb) * FRAG), accE[nb]);
    }

    // ---- GEMM 2: lp = log(conc+eps) @ s_neg  ([16,128] x [128,64]) ----
    v8f accL[4];
#pragma unroll
    for (int nb = 0; nb < 4; ++nb) accL[nb] = zero8();
#pragma unroll
    for (int kblk = 0; kblk < KB_M - 1; ++kblk) {    // K 0..95: unconditional
      const v16h a = a_frag_log_f2(c_row, kblk * 32, h_a);
#pragma unroll
      for (int nb = 0; nb < 4; ++nb)
        accL[nb] = WMMA_F16(a, frag_sw(sneg_sw + (kblk * 4 + nb) * FRAG), accL[nb]);
    }
    {                                                // K 96..127: clamped tail
      const v16h a = a_frag_log_clamp(c_row, (KB_M - 1) * 32, h_a, N_METS);
#pragma unroll
      for (int nb = 0; nb < 4; ++nb)
        accL[nb] = WMMA_F16(a, frag_sw(sneg_sw + ((KB_M - 1) * 4 + nb) * FRAG), accL[nb]);
    }

    // ---- rates = kcat * sigmoid(enzyme) * exp(lp) ----
    // write straight into the GEMM-3 A-fragment swizzled layout
#pragma unroll
    for (int nb = 0; nb < 4; ++nb) {
      const int j = nb * 16 + col;                // reaction index
      const float kc = kcat_lds[j];
      const int kblk = j >> 5;
      const int o = j & 31;
      const int hh = (o >> 3) & 1;
      const int ii = (o & 7) + ((o >> 4) << 3);
#pragma unroll
      for (int r = 0; r < 8; ++r) {
        const int m = r + row_off;
        const float sig  = 1.0f / (1.0f + __expf(-accE[nb][r]));
        const float rate = kc * sig * __expf(accL[nb][r]);
        rs[kblk * FRAG + (m + 16 * hh) * 16 + ii] = (_Float16)rate;
      }
    }
    // per-wave LDS ops are in-order; compiler inserts the dscnt wait for the RAW

    // ---- GEMM 3: d_conc = rates @ S^T  ([16,64] x [64,128]) ----
    v8f accD[8];
#pragma unroll
    for (int nb = 0; nb < 8; ++nb) accD[nb] = zero8();
#pragma unroll
    for (int kblk = 0; kblk < KB_R; ++kblk) {
      const v16h a = frag_sw(rs + kblk * FRAG);
#pragma unroll
      for (int nb = 0; nb < 8; ++nb)
        accD[nb] = WMMA_F16(a, frag_sw(st_sw + (kblk * 8 + nb) * FRAG), accD[nb]);
    }

    // ---- Euler step: out = max(conc + DT * d_conc, 0) ----
#pragma unroll
    for (int nb = 0; nb < 8; ++nb) {
      const int i = nb * 16 + col;
      if (i < N_METS) {
#pragma unroll
        for (int r = 0; r < 8; ++r) {
          const int m = r + row_off;
          const size_t off = (cell0 + m) * (size_t)N_METS + i;
          out[off] = fmaxf(conc[off] + DT_STEP * accD[nb][r], 0.0f);
        }
      }
    }
  }
}

extern "C" void kernel_launch(void* const* d_in, const int* in_sizes, int n_in,
                              void* d_out, int out_size, void* d_ws, size_t ws_size,
                              hipStream_t stream) {
  const float* conc      = (const float*)d_in[0];
  const float* gene_expr = (const float*)d_in[1];
  const float* S         = (const float*)d_in[2];
  const float* G         = (const float*)d_in[3];
  const float* kcat      = (const float*)d_in[4];
  float* out = (float*)d_out;

  const int Bcells  = in_sizes[0] / N_METS;   // conc is [B, N_METS]
  const int n_tiles = (Bcells + 15) / 16;
  int blocks = (n_tiles + WAVES_PER_BLOCK - 1) / WAVES_PER_BLOCK;
  if (blocks > 1024) blocks = 1024;
  syn3a_kinetics_step<<<blocks, BLOCK_THREADS, 0, stream>>>(
      conc, gene_expr, S, G, kcat, out, n_tiles);
}
